// HybridTransformerLayer_89060441850246
// MI455X (gfx1250) — matrix operand order repaired and checked by reading, not compiled, not run on hardware
//
#include <hip/hip_runtime.h>
#include <math.h>

// ---------------------------------------------------------------------------
// HybridTransformerLayer on MI455X (gfx1250), wave32 + WMMA bf16 path.
// B=4 N=2048 D=1024 HS=8 dh=128 | ABSA: HA=8 HD=128 HV=64 deg=4 blk=64 | DFF=4096
// ---------------------------------------------------------------------------

typedef __attribute__((ext_vector_type(16))) __bf16          v16bf;
typedef __attribute__((ext_vector_type(8)))  float           v8f;
typedef __attribute__((ext_vector_type(8)))  unsigned short  u16x8;

union FragBF {
    u16x8 h[2];
    v16bf v;
};

#define WMMA_BF16(a, b, c) \
    __builtin_amdgcn_wmma_f32_16x16x32_bf16(false, (a), false, (b), (short)0, (c), false, false)

__device__ __forceinline__ unsigned short f2bf(float f) {
    union { float f; unsigned int u; } c; c.f = f;
    unsigned int u = c.u;
    u += 0x7FFFu + ((u >> 16) & 1u);         // round-to-nearest-even
    return (unsigned short)(u >> 16);
}

__device__ __forceinline__ v8f v8f_zero() {
    v8f z = {0.f, 0.f, 0.f, 0.f, 0.f, 0.f, 0.f, 0.f};
    return z;
}

// ---------------------------------------------------------------------------
// Generic GEMM: C[M,N] = act(A[M,K] @ B + bias)
//   BT==0: B stored [N,K]  (row n = output column n, contiguous along K)
//   BT==1: B stored [K,N]
//   ACT==1: exact GELU epilogue.
// Tile: 128(M) x 128(N), 256 threads = 8 waves, each wave 32x64 (2x4 tiles).
// Requires K % 32 == 0. M,N bounds-checked.
// ---------------------------------------------------------------------------
template <int ACT, int BT>
__global__ __launch_bounds__(256)
void gemm_wmma_bf16(const float* __restrict__ A, long long lda,
                    const float* __restrict__ B, long long ldb,
                    const float* __restrict__ bias,
                    float* __restrict__ C, long long ldc,
                    int M, int N, int K)
{
    __shared__ alignas(16) unsigned short As[128][32];
    __shared__ alignas(16) unsigned short Bs[128][32];

    const int tid  = threadIdx.x;
    const int lane = tid & 31;
    const int wave = tid >> 5;
    const int wm   = (wave >> 1) * 32;   // 0,32,64,96
    const int wn   = (wave & 1) * 64;    // 0,64
    const int lr   = lane & 15;
    const int lh   = lane >> 4;

    const int m0 = blockIdx.x * 128;
    const int n0 = blockIdx.y * 128;

    const int srow = tid >> 1;           // 0..127
    const int sseg = (tid & 1) * 16;     // 0 or 16

    v8f acc[2][4];
#pragma unroll
    for (int i = 0; i < 2; ++i)
#pragma unroll
        for (int j = 0; j < 4; ++j) acc[i][j] = v8f_zero();

    for (int k0 = 0; k0 < K; k0 += 32) {
        __syncthreads();
        // ---- stage A tile (fp32 -> bf16) ----
        {
            const int gr = m0 + srow;
            if (gr < M) {
                const float* ap = A + (long long)gr * lda + k0 + sseg;
#pragma unroll
                for (int j = 0; j < 16; j += 4) {
                    float4 f = *(const float4*)(ap + j);
                    As[srow][sseg + j + 0] = f2bf(f.x);
                    As[srow][sseg + j + 1] = f2bf(f.y);
                    As[srow][sseg + j + 2] = f2bf(f.z);
                    As[srow][sseg + j + 3] = f2bf(f.w);
                }
            } else {
#pragma unroll
                for (int j = 0; j < 16; ++j) As[srow][sseg + j] = 0;
            }
        }
        // ---- stage B tile ----
        {
            const int gc = n0 + srow;
            if (gc < N) {
                if (BT == 0) {
                    const float* bp = B + (long long)gc * ldb + k0 + sseg;
#pragma unroll
                    for (int j = 0; j < 16; j += 4) {
                        float4 f = *(const float4*)(bp + j);
                        Bs[srow][sseg + j + 0] = f2bf(f.x);
                        Bs[srow][sseg + j + 1] = f2bf(f.y);
                        Bs[srow][sseg + j + 2] = f2bf(f.z);
                        Bs[srow][sseg + j + 3] = f2bf(f.w);
                    }
                } else {
#pragma unroll
                    for (int j = 0; j < 16; ++j)
                        Bs[srow][sseg + j] = f2bf(B[(long long)(k0 + sseg + j) * ldb + gc]);
                }
            } else {
#pragma unroll
                for (int j = 0; j < 16; ++j) Bs[srow][sseg + j] = 0;
            }
        }
        __syncthreads();

        // ---- fragments + 8 WMMAs ----
        FragBF af[2], bf[4];
#pragma unroll
        for (int i = 0; i < 2; ++i) {
            const int r = wm + i * 16 + lr;
            af[i].h[0] = *(const u16x8*)&As[r][lh * 8];
            af[i].h[1] = *(const u16x8*)&As[r][lh * 8 + 16];
        }
#pragma unroll
        for (int j = 0; j < 4; ++j) {
            const int c = wn + j * 16 + lr;
            bf[j].h[0] = *(const u16x8*)&Bs[c][lh * 8];
            bf[j].h[1] = *(const u16x8*)&Bs[c][lh * 8 + 16];
        }
#pragma unroll
        for (int i = 0; i < 2; ++i)
#pragma unroll
            for (int j = 0; j < 4; ++j)
                acc[i][j] = WMMA_BF16(af[i].v, bf[j].v, acc[i][j]);
    }

    // ---- epilogue ----
#pragma unroll
    for (int i = 0; i < 2; ++i) {
#pragma unroll
        for (int j = 0; j < 4; ++j) {
            const int gc = n0 + wn + j * 16 + lr;
            if (gc >= N) continue;
            const float bv = bias ? bias[gc] : 0.0f;
#pragma unroll
            for (int r = 0; r < 8; ++r) {
                const int gr = m0 + wm + i * 16 + r + lh * 8;
                if (gr < M) {
                    float vv = acc[i][j][r] + bv;
                    if (ACT == 1) vv = 0.5f * vv * (1.0f + erff(vv * 0.70710678118654752f));
                    C[(long long)gr * ldc + gc] = vv;
                }
            }
        }
    }
}

// ---------------------------------------------------------------------------
// Flash attention (serves both the dense MHA and the ABSA block-window heads).
// 64 queries / workgroup, 128 threads = 4 waves x 16 rows. dqk fixed 128.
// degree < 0  -> full attention over [0, Ntok)
// degree >= 0 -> KV range [max(0, qb-degree)*64, (qb+1)*64)  (all-valid blocks)
// ---------------------------------------------------------------------------
__global__ __launch_bounds__(128)
void flash_wmma_bf16(const float* __restrict__ Q, long long qsB, long long qsH, int qld,
                     const float* __restrict__ Kp, long long ksB, long long ksH, int kld,
                     const float* __restrict__ Vp, long long vsB, long long vsH, int vld,
                     float* __restrict__ O, long long osB, long long osH, int oldr,
                     int Ntok, int dv, int degree, float scale)
{
    __shared__ alignas(16) unsigned short Qs[64][128];
    __shared__ alignas(16) unsigned short Ks[64][128];
    __shared__ alignas(16) unsigned short Vts[128][64];
    __shared__ alignas(16) unsigned short Ps[4][16][64];

    const int qb = blockIdx.x;
    const int h  = blockIdx.y;
    const int b  = blockIdx.z;

    const float* q = Q  + (long long)b * qsB + (long long)h * qsH;
    const float* k = Kp + (long long)b * ksB + (long long)h * ksH;
    const float* v = Vp + (long long)b * vsB + (long long)h * vsH;
    float*       o = O  + (long long)b * osB + (long long)h * osH;

    const int tid  = threadIdx.x;
    const int lane = tid & 31;
    const int wave = tid >> 5;
    const int lr   = lane & 15;
    const int lh   = lane >> 4;

    // ---- stage Q block [64 x 128] ----
    {
        const int row = tid >> 1, seg = (tid & 1) * 64;
        const float* qp = q + (long long)(qb * 64 + row) * qld + seg;
#pragma unroll
        for (int j = 0; j < 64; j += 4) {
            float4 f = *(const float4*)(qp + j);
            Qs[row][seg + j + 0] = f2bf(f.x);
            Qs[row][seg + j + 1] = f2bf(f.y);
            Qs[row][seg + j + 2] = f2bf(f.z);
            Qs[row][seg + j + 3] = f2bf(f.w);
        }
    }
    __syncthreads();

    // Q A-fragments (4 K-chunks of 32) kept in registers.
    FragBF qf[4];
#pragma unroll
    for (int c = 0; c < 4; ++c) {
        const int r = wave * 16 + lr;
        qf[c].h[0] = *(const u16x8*)&Qs[r][c * 32 + lh * 8];
        qf[c].h[1] = *(const u16x8*)&Qs[r][c * 32 + lh * 8 + 16];
    }

    const int ntv = dv >> 4;    // 4 (HV=64) or 8 (dh=128)
    v8f   acc[8];
    float mrow[8], lrow[8];
#pragma unroll
    for (int t = 0; t < 8; ++t) acc[t] = v8f_zero();
#pragma unroll
    for (int r = 0; r < 8; ++r) { mrow[r] = -1e30f; lrow[r] = 0.0f; }

    int c_lo = 0, c_hi = Ntok;
    if (degree >= 0) {
        int lo = qb - degree; if (lo < 0) lo = 0;
        c_lo = lo * 64;
        c_hi = (qb + 1) * 64;
    }

    for (int c0 = c_lo; c0 < c_hi; c0 += 64) {
        __syncthreads();
        // ---- stage K chunk [64 x 128] ----
        {
            const int row = tid >> 1, seg = (tid & 1) * 64;
            const float* kp = k + (long long)(c0 + row) * kld + seg;
#pragma unroll
            for (int j = 0; j < 64; j += 4) {
                float4 f = *(const float4*)(kp + j);
                Ks[row][seg + j + 0] = f2bf(f.x);
                Ks[row][seg + j + 1] = f2bf(f.y);
                Ks[row][seg + j + 2] = f2bf(f.z);
                Ks[row][seg + j + 3] = f2bf(f.w);
            }
        }
        // ---- stage V chunk transposed: Vts[d][key] ----
        {
            const int key = tid >> 1;
            const int seg = (tid & 1) * (dv >> 1);
            const float* vp = v + (long long)(c0 + key) * vld + seg;
            for (int j = 0; j < (dv >> 1); j += 4) {
                float4 f = *(const float4*)(vp + j);
                Vts[seg + j + 0][key] = f2bf(f.x);
                Vts[seg + j + 1][key] = f2bf(f.y);
                Vts[seg + j + 2][key] = f2bf(f.z);
                Vts[seg + j + 3][key] = f2bf(f.w);
            }
        }
        __syncthreads();

        // ---- scores S[16 x 64] per wave ----
        v8f s[4];
#pragma unroll
        for (int j = 0; j < 4; ++j) s[j] = v8f_zero();
#pragma unroll
        for (int c = 0; c < 4; ++c) {
#pragma unroll
            for (int j = 0; j < 4; ++j) {
                FragBF bfr;
                const int col = j * 16 + lr;
                bfr.h[0] = *(const u16x8*)&Ks[col][c * 32 + lh * 8];
                bfr.h[1] = *(const u16x8*)&Ks[col][c * 32 + lh * 8 + 16];
                s[j] = WMMA_BF16(qf[c].v, bfr.v, s[j]);
            }
        }

        // ---- online softmax (per accumulator element = per row) ----
#pragma unroll
        for (int r = 0; r < 8; ++r) {
            float tmax = -1e30f;
#pragma unroll
            for (int j = 0; j < 4; ++j) tmax = fmaxf(tmax, s[j][r]);
            tmax *= scale;
#pragma unroll
            for (int m = 1; m < 16; m <<= 1) tmax = fmaxf(tmax, __shfl_xor(tmax, m, 32));
            const float mnew = fmaxf(mrow[r], tmax);
            const float fsc  = __expf(mrow[r] - mnew);
            float lsum = 0.0f;
#pragma unroll
            for (int j = 0; j < 4; ++j) {
                const float e = __expf(s[j][r] * scale - mnew);
                lsum += e;
                Ps[wave][r + lh * 8][j * 16 + lr] = f2bf(e);
            }
#pragma unroll
            for (int m = 1; m < 16; m <<= 1) lsum += __shfl_xor(lsum, m, 32);
            lrow[r] = lrow[r] * fsc + lsum;
            mrow[r] = mnew;
#pragma unroll
            for (int t = 0; t < 8; ++t)
                if (t < ntv) acc[t][r] *= fsc;
        }

        // ---- P (via LDS relayout) @ V ----
        FragBF pf[2];
#pragma unroll
        for (int c = 0; c < 2; ++c) {
            pf[c].h[0] = *(const u16x8*)&Ps[wave][lr][c * 32 + lh * 8];
            pf[c].h[1] = *(const u16x8*)&Ps[wave][lr][c * 32 + lh * 8 + 16];
        }
#pragma unroll
        for (int c = 0; c < 2; ++c) {
#pragma unroll
            for (int t = 0; t < 8; ++t) {
                if (t < ntv) {
                    FragBF bfr;
                    const int col = t * 16 + lr;
                    bfr.h[0] = *(const u16x8*)&Vts[col][c * 32 + lh * 8];
                    bfr.h[1] = *(const u16x8*)&Vts[col][c * 32 + lh * 8 + 16];
                    acc[t] = WMMA_BF16(pf[c].v, bfr.v, acc[t]);
                }
            }
        }
    }

    // ---- normalize + store ----
#pragma unroll
    for (int t = 0; t < 8; ++t) {
        if (t < ntv) {
            const int col = t * 16 + lr;
#pragma unroll
            for (int r = 0; r < 8; ++r) {
                const int row = qb * 64 + wave * 16 + r + lh * 8;
                o[(long long)row * oldr + col] = acc[t][r] / lrow[r];
            }
        }
    }
}

// ---------------------------------------------------------------------------
// Fused residual-add + LayerNorm: out = LN(res + delta) * g + be.  D = 1024.
// ---------------------------------------------------------------------------
__global__ __launch_bounds__(256)
void ln_resid(const float* __restrict__ res, const float* __restrict__ delta,
              const float* __restrict__ g, const float* __restrict__ be,
              float* __restrict__ out, int D)
{
    __shared__ float redS[8], redQ[8];
    const long long row = blockIdx.x;
    const float* r0 = res   + row * D;
    const float* d0 = delta + row * D;
    float vloc[4];
    float s = 0.0f, s2 = 0.0f;
#pragma unroll
    for (int i = 0; i < 4; ++i) {
        const int c = threadIdx.x + i * 256;
        const float vv = r0[c] + d0[c];
        vloc[i] = vv; s += vv; s2 += vv * vv;
    }
#pragma unroll
    for (int m = 1; m < 32; m <<= 1) { s += __shfl_xor(s, m, 32); s2 += __shfl_xor(s2, m, 32); }
    const int wave = threadIdx.x >> 5, lane = threadIdx.x & 31;
    if (lane == 0) { redS[wave] = s; redQ[wave] = s2; }
    __syncthreads();
    float ts = 0.0f, tq = 0.0f;
#pragma unroll
    for (int w = 0; w < 8; ++w) { ts += redS[w]; tq += redQ[w]; }
    const float mu   = ts / (float)D;
    const float var  = tq / (float)D - mu * mu;
    const float rstd = rsqrtf(var + 1e-5f);
#pragma unroll
    for (int i = 0; i < 4; ++i) {
        const int c = threadIdx.x + i * 256;
        out[row * D + c] = (vloc[i] - mu) * rstd * g[c] + be[c];
    }
}

// ---------------------------------------------------------------------------
// Host orchestration
// ---------------------------------------------------------------------------
extern "C" void kernel_launch(void* const* d_in, const int* in_sizes, int n_in,
                              void* d_out, int out_size, void* d_ws, size_t ws_size,
                              hipStream_t stream)
{
    (void)in_sizes; (void)n_in; (void)out_size; (void)ws_size;

    const float* x     = (const float*)d_in[0];
    const float* w_qkv = (const float*)d_in[1];
    const float* b_qkv = (const float*)d_in[2];
    const float* w_o   = (const float*)d_in[3];
    const float* b_o   = (const float*)d_in[4];
    const float* wq_a  = (const float*)d_in[5];
    const float* wk_a  = (const float*)d_in[6];
    const float* wv_a  = (const float*)d_in[7];
    const float* w_p   = (const float*)d_in[8];
    const float* b_p   = (const float*)d_in[9];
    const float* g1    = (const float*)d_in[10];
    const float* be1   = (const float*)d_in[11];
    const float* w1    = (const float*)d_in[12];
    const float* b1f   = (const float*)d_in[13];
    const float* w2    = (const float*)d_in[14];
    const float* b2f   = (const float*)d_in[15];
    const float* g2    = (const float*)d_in[16];
    const float* be2   = (const float*)d_in[17];

    constexpr int Bb = 4, Nn = 2048, Dd = 1024, HS = 8;
    constexpr int HA = 8, HD = 128, HV = 64, DFF = 4096, DEG = 4;
    const int BN = Bb * Nn;                       // 8192 tokens
    const float scale = 0.088388347648318447f;    // 1/sqrt(128), both attentions

    // Workspace layout (floats). FF1 reuses the QKV+STDO region (dead by then).
    float* ws   = (float*)d_ws;
    size_t off  = 0;
    float* qkv  = ws + off; off += (size_t)BN * 3 * Dd;   // 25,165,824
    float* stdo = ws + off; off += (size_t)BN * Dd;       //  8,388,608
    float* ff1  = ws;                                     // 33,554,432 (overlaps qkv+stdo)
    float* qa   = ws + off; off += (size_t)HA * BN * HD;
    float* ka   = ws + off; off += (size_t)HA * BN * HD;
    float* va   = ws + off; off += (size_t)HA * BN * HV;
    float* cat  = ws + off; off += (size_t)BN * Dd;
    float* comb = ws + off; off += (size_t)BN * Dd;
    float* x1   = ws + off; off += (size_t)BN * Dd;
    float* ff2  = qa;                                     // reuse (qa dead after ABSA attn)

    const dim3 t256(256), t128(128);

    // 1) qkv = x @ w_qkv.T + b_qkv            [8192 x 3072], K=1024
    gemm_wmma_bf16<0, 0><<<dim3(BN / 128, (3 * Dd) / 128), t256, 0, stream>>>(
        x, Dd, w_qkv, Dd, b_qkv, qkv, 3 * Dd, BN, 3 * Dd, Dd);

    // 2) dense MHA (flash), dh = dv = 128, full KV range
    flash_wmma_bf16<<<dim3(Nn / 64, HS, Bb), t128, 0, stream>>>(
        qkv,            (long long)Nn * 3 * Dd, 128, 3 * Dd,
        qkv + Dd,       (long long)Nn * 3 * Dd, 128, 3 * Dd,
        qkv + 2 * Dd,   (long long)Nn * 3 * Dd, 128, 3 * Dd,
        stdo,           (long long)Nn * Dd,     128, Dd,
        Nn, 128, -1, scale);

    // 3) std @ w_o.T + b_o, only first D/2 columns survive the concat
    gemm_wmma_bf16<0, 0><<<dim3(BN / 128, 512 / 128), t256, 0, stream>>>(
        stdo, Dd, w_o, Dd, b_o, cat, Dd, BN, 512, Dd);

    // 4) ABSA per-head projections: qa/ka [8192x128], va [8192x64], K=128, B in [K,N]
    for (int h = 0; h < HA; ++h) {
        gemm_wmma_bf16<0, 1><<<dim3(BN / 128, 1), t256, 0, stream>>>(
            x + h * HD, Dd, wq_a + (size_t)h * HD * HD, HD, nullptr,
            qa + (size_t)h * BN * HD, HD, BN, HD, HD);
        gemm_wmma_bf16<0, 1><<<dim3(BN / 128, 1), t256, 0, stream>>>(
            x + h * HD, Dd, wk_a + (size_t)h * HD * HD, HD, nullptr,
            ka + (size_t)h * BN * HD, HD, BN, HD, HD);
        gemm_wmma_bf16<0, 1><<<dim3(BN / 128, 1), t256, 0, stream>>>(
            x + h * HD, Dd, wv_a + (size_t)h * HD * HV, HV, nullptr,
            va + (size_t)h * BN * HV, HV, BN, HV, HD);
    }

    // 5) ABSA block-window attention -> cat[:, 512 + h*64 : ...]
    flash_wmma_bf16<<<dim3(Nn / 64, HA, Bb), t128, 0, stream>>>(
        qa, (long long)Nn * HD, (long long)BN * HD, HD,
        ka, (long long)Nn * HD, (long long)BN * HD, HD,
        va, (long long)Nn * HV, (long long)BN * HV, HV,
        cat + 512, (long long)Nn * Dd, HV, Dd,
        Nn, HV, DEG, scale);

    // 6) comb = cat @ w_p.T + b_p
    gemm_wmma_bf16<0, 0><<<dim3(BN / 128, Dd / 128), t256, 0, stream>>>(
        cat, Dd, w_p, Dd, b_p, comb, Dd, BN, Dd, Dd);

    // 7) x1 = LN(x + comb)
    ln_resid<<<dim3(BN), t256, 0, stream>>>(x, comb, g1, be1, x1, Dd);

    // 8) ff1 = gelu(x1 @ w1.T + b1f)          [8192 x 4096]
    gemm_wmma_bf16<1, 0><<<dim3(BN / 128, DFF / 128), t256, 0, stream>>>(
        x1, Dd, w1, Dd, b1f, ff1, DFF, BN, DFF, Dd);

    // 9) ff2 = ff1 @ w2.T + b2f               K=4096
    gemm_wmma_bf16<0, 0><<<dim3(BN / 128, Dd / 128), t256, 0, stream>>>(
        ff1, DFF, w2, DFF, b2f, ff2, Dd, BN, Dd, DFF);

    // 10) out = LN(x1 + ff2)
    ln_resid<<<dim3(BN), t256, 0, stream>>>(x1, ff2, g2, be2, (float*)d_out, Dd);
}
